// TransformerLayer_17652315587090
// MI455X (gfx1250) — matrix operand order repaired and checked by reading, not compiled, not run on hardware
//
#include <hip/hip_runtime.h>
#include <hip/hip_bf16.h>
#include <math.h>

// ---------------------------------------------------------------------------
// Types for CDNA5 WMMA (wave32): v_wmma_f32_16x16x32_bf16
// ---------------------------------------------------------------------------
typedef __bf16 bf16_t;
typedef __attribute__((ext_vector_type(8)))  float  v8f;
typedef __attribute__((ext_vector_type(8)))  bf16_t v8bf;
typedef __attribute__((ext_vector_type(16))) bf16_t v16bf;

#define DEV __device__ __forceinline__

// Problem constants (B=2, T=2048, D=1024, H=16, dh=64, F=4096, K=3)
#define BT    4096        // B*T
#define TT    2048
#define DD    1024
#define HH    16
#define DH    64
#define FF    4096
#define PADT  2050        // T + 2 (causal conv left pad rows)
#define QKV_STRIDE 4194304ll   // BT*DD

DEV v16bf pack16(const bf16_t* p0, const bf16_t* p1) {
  v8bf lo = *(const v8bf*)p0;
  v8bf hi = *(const v8bf*)p1;
  v16bf r;
#pragma unroll
  for (int i = 0; i < 8; ++i) { r[i] = lo[i]; r[i + 8] = hi[i]; }
  return r;
}

DEV v8f wmma_bf16(v16bf a, v16bf b, v8f c) {
  // (neg_a, A, neg_b, B, c_mod, C, reuse_a, reuse_b)
  return __builtin_amdgcn_wmma_f32_16x16x32_bf16(false, a, false, b, (short)0, c, false, false);
}

DEV float gelu_tanh(float x) {
  float t = 0.7978845608028654f * (x + 0.044715f * x * x * x);
  return 0.5f * x * (1.0f + tanhf(t));
}

// ---------------------------------------------------------------------------
// fp32 -> bf16 convert (weights)
// ---------------------------------------------------------------------------
__global__ void cvt_kernel(const float* __restrict__ in, bf16_t* __restrict__ out, long n) {
  long i = (long)blockIdx.x * 256 + threadIdx.x;
  if (i < n) out[i] = (bf16_t)in[i];
}

// conv weight repack: in [Co][Ci][3] fp32 -> out [Co][i*Ci + ci] bf16 (tap-major)
__global__ void repack_kernel(const float* __restrict__ in, bf16_t* __restrict__ out,
                              int Ci, long total) {
  long idx = (long)blockIdx.x * 256 + threadIdx.x;
  if (idx >= total) return;
  int i  = (int)(idx % 3);
  long r = idx / 3;
  int ci = (int)(r % Ci);
  long co = r / Ci;
  out[co * 3 * (long)Ci + (long)i * Ci + ci] = (bf16_t)in[idx];
}

__global__ void zero_bf16_kernel(bf16_t* __restrict__ p, int n) {
  int i = blockIdx.x * 256 + threadIdx.x;
  if (i < n) p[i] = (bf16_t)0.0f;
}

// ---------------------------------------------------------------------------
// LayerNorm over D=1024 (one block per row, 256 threads).
// mode 0: out_bf16[row] = LN(x)*g+b + pos[t]      (dense rows)
// mode 1: out_bf16 written into padded buffer at row (b*(T+2)+2+t)
// ---------------------------------------------------------------------------
__global__ __launch_bounds__(256) void ln_kernel(const float* __restrict__ x,
                                                 const float* __restrict__ g,
                                                 const float* __restrict__ bta,
                                                 const float* __restrict__ pos,
                                                 bf16_t* __restrict__ out, int mode) {
  int row = blockIdx.x;            // 0..BT-1
  int tid = threadIdx.x;
  const float* xr = x + (long)row * DD;
  float4 vx = ((const float4*)xr)[tid];
  float s  = vx.x + vx.y + vx.z + vx.w;
  float s2 = vx.x * vx.x + vx.y * vx.y + vx.z * vx.z + vx.w * vx.w;
#pragma unroll
  for (int off = 16; off > 0; off >>= 1) {
    s  += __shfl_xor(s,  off, 32);
    s2 += __shfl_xor(s2, off, 32);
  }
  __shared__ float red0[8], red1[8];
  int wv = tid >> 5, ln = tid & 31;
  if (ln == 0) { red0[wv] = s; red1[wv] = s2; }
  __syncthreads();
  if (tid == 0) {
    float a = 0.f, c = 0.f;
#pragma unroll
    for (int i = 0; i < 8; ++i) { a += red0[i]; c += red1[i]; }
    red0[0] = a; red1[0] = c;
  }
  __syncthreads();
  float mu   = red0[0] * (1.0f / DD);
  float var  = red1[0] * (1.0f / DD) - mu * mu;
  float rstd = rsqrtf(var + 1e-5f);

  int t = row % TT;
  bf16_t* orow;
  if (mode == 0) orow = out + (long)row * DD;
  else {
    int b = row / TT;
    orow = out + ((long)b * PADT + 2 + t) * DD;
  }
  float4 gg = ((const float4*)g)[tid];
  float4 bb = ((const float4*)bta)[tid];
  float4 pp = make_float4(0.f, 0.f, 0.f, 0.f);
  if (mode == 0) pp = ((const float4*)(pos + (long)t * DD))[tid];
  int c0 = tid * 4;
  orow[c0 + 0] = (bf16_t)((vx.x - mu) * rstd * gg.x + bb.x + pp.x);
  orow[c0 + 1] = (bf16_t)((vx.y - mu) * rstd * gg.y + bb.y + pp.y);
  orow[c0 + 2] = (bf16_t)((vx.z - mu) * rstd * gg.z + bb.z + pp.z);
  orow[c0 + 3] = (bf16_t)((vx.w - mu) * rstd * gg.w + bb.w + pp.w);
}

// ---------------------------------------------------------------------------
// Generic bf16 WMMA GEMM:  C[M,N] = A[M,:K] * B^T   (B stored [N][K] row-major)
// A row m starts at A + m*lda (lda may be < K: causal-conv window trick).
// Block: 256 threads = 8 waves; block tile 128x128, BK=32, LDS double-buffered.
// Wave (wm 0..3, wn 0..1) owns a 32x64 tile: 2 A-frags x 4 B-frags = 8 WMMA/step.
// mode 0: QKV scatter  -> outb[which][m][col] bf16   (which = n>>10)
// mode 1: residual add -> outf[m*1024+n] = resid[m*1024+n] + acc   (N==1024)
// mode 2: GELU bf16    -> outb[m*ostride + n]
// ---------------------------------------------------------------------------
__global__ __launch_bounds__(256) void gemm_bf16_kernel(
    const bf16_t* __restrict__ A, long lda,
    const bf16_t* __restrict__ Bw, int K,
    int mode, const float* __restrict__ resid,
    float* __restrict__ outf, bf16_t* __restrict__ outb, long ostride) {
  __shared__ bf16_t As[2][128][40];
  __shared__ bf16_t Bs[2][128][40];

  int tid  = threadIdx.x;
  int lane = tid & 31;
  int w    = tid >> 5;
  int wm   = w & 3;        // 4 row subtiles of 32
  int wn   = w >> 2;       // 2 col subtiles of 64
  int grp  = lane >> 4;    // 0/1 half-wave
  int l16  = lane & 15;

  long tileM = (long)blockIdx.y * 128;
  long tileN = (long)blockIdx.x * 128;

  v8f acc[2][4];
  v8f vz = {};
#pragma unroll
  for (int mt = 0; mt < 2; ++mt)
#pragma unroll
    for (int nt = 0; nt < 4; ++nt) acc[mt][nt] = vz;

  int ldr = tid >> 1;            // 0..127
  int ldc = (tid & 1) * 16;      // 0 or 16

  const bf16_t* Aptr = A  + (tileM + ldr) * lda + ldc;
  const bf16_t* Bptr = Bw + (tileN + ldr) * (long)K + ldc;

  // preload K-tile 0 into buffer 0
  {
    v8bf a0 = *(const v8bf*)(Aptr);
    v8bf a1 = *(const v8bf*)(Aptr + 8);
    v8bf b0 = *(const v8bf*)(Bptr);
    v8bf b1 = *(const v8bf*)(Bptr + 8);
    *(v8bf*)&As[0][ldr][ldc]     = a0;
    *(v8bf*)&As[0][ldr][ldc + 8] = a1;
    *(v8bf*)&Bs[0][ldr][ldc]     = b0;
    *(v8bf*)&Bs[0][ldr][ldc + 8] = b1;
  }
  __syncthreads();

  int buf = 0;
  for (int k0 = 0; k0 < K; k0 += 32) {
    int nxt = k0 + 32;
    bool have = nxt < K;
    v8bf na0 = {}, na1 = {}, nb0 = {}, nb1 = {};
    if (have) {
      // issue next tile's global loads early (latency hidden under WMMAs)
      na0 = *(const v8bf*)(Aptr + nxt);
      na1 = *(const v8bf*)(Aptr + nxt + 8);
      nb0 = *(const v8bf*)(Bptr + nxt);
      nb1 = *(const v8bf*)(Bptr + nxt + 8);
      if (nxt + 32 < K) {
        __builtin_prefetch(Aptr + nxt + 32, 0, 0);  // global_prefetch_b8
        __builtin_prefetch(Bptr + nxt + 32, 0, 0);
      }
    }

    // compute on current buffer: 2 A-frags x 4 B-frags -> 8 WMMA
    v16bf af[2], bfq[4];
#pragma unroll
    for (int mt = 0; mt < 2; ++mt) {
      const bf16_t* ar = &As[buf][wm * 32 + mt * 16 + l16][0];
      af[mt] = pack16(ar + grp * 8, ar + 16 + grp * 8);
    }
#pragma unroll
    for (int nt = 0; nt < 4; ++nt) {
      const bf16_t* br = &Bs[buf][wn * 64 + nt * 16 + l16][0];
      bfq[nt] = pack16(br + grp * 8, br + 16 + grp * 8);
    }
#pragma unroll
    for (int mt = 0; mt < 2; ++mt)
#pragma unroll
      for (int nt = 0; nt < 4; ++nt)
        acc[mt][nt] = wmma_bf16(af[mt], bfq[nt], acc[mt][nt]);

    if (have) {
      *(v8bf*)&As[buf ^ 1][ldr][ldc]     = na0;
      *(v8bf*)&As[buf ^ 1][ldr][ldc + 8] = na1;
      *(v8bf*)&Bs[buf ^ 1][ldr][ldc]     = nb0;
      *(v8bf*)&Bs[buf ^ 1][ldr][ldc + 8] = nb1;
    }
    __syncthreads();
    buf ^= 1;
  }

#pragma unroll
  for (int mt = 0; mt < 2; ++mt)
#pragma unroll
    for (int nt = 0; nt < 4; ++nt) {
      long ncol = tileN + wn * 64 + nt * 16 + l16;
#pragma unroll
      for (int r = 0; r < 8; ++r) {
        long m = tileM + wm * 32 + mt * 16 + grp * 8 + r;
        float val = acc[mt][nt][r];
        if (mode == 0) {
          long which = ncol >> 10;
          long col   = ncol & 1023;
          outb[which * QKV_STRIDE + m * DD + col] = (bf16_t)val;
        } else if (mode == 1) {
          long o = m * DD + ncol;
          outf[o] = resid[o] + val;
        } else {
          outb[m * ostride + ncol] = (bf16_t)gelu_tanh(val);
        }
      }
    }
}

// ---------------------------------------------------------------------------
// Flash attention, bf16 WMMA, wave32.
// Grid: (T/64, H, B).  Block: 128 threads = 4 waves; wave owns 16 query rows.
// Q/K/V: bf16 [B*T][1024], head h occupies cols h*64..h*64+63.
// Reference masks padded QUERY rows (uniform softmax then zeroed) -> we apply
// causal mask in-score and write 0 for padded query rows.
// ---------------------------------------------------------------------------
__global__ __launch_bounds__(128) void attn_kernel(const bf16_t* __restrict__ Q,
                                                   const bf16_t* __restrict__ Kb,
                                                   const bf16_t* __restrict__ Vb,
                                                   const unsigned char* __restrict__ msk,
                                                   bf16_t* __restrict__ Y) {
  __shared__ bf16_t VT[64][72];       // transposed V tile: VT[d][s]
  __shared__ bf16_t Pm[4][16][72];    // per-wave probabilities

  int qt = blockIdx.x, h = blockIdx.y, b = blockIdx.z;
  int tid = threadIdx.x, lane = tid & 31, w = tid >> 5;
  int grp = lane >> 4, l16 = lane & 15;

  long baseBT = (long)b * TT;
  long hoff   = (long)h * DH;
  int  qbase  = qt * 64 + w * 16;

  // Q A-fragments (dh=64 -> two 16x32 fragments), kept in registers
  v16bf aQ[2];
  {
    const bf16_t* qrow = Q + (baseBT + qbase + l16) * DD + hoff;
#pragma unroll
    for (int kt = 0; kt < 2; ++kt)
      aQ[kt] = pack16(qrow + kt * 32 + grp * 8, qrow + kt * 32 + 16 + grp * 8);
  }

  float mrow[8], lrow[8];
  v8f accO[4];
#pragma unroll
  for (int r = 0; r < 8; ++r) { mrow[r] = -3.0e38f; lrow[r] = 0.f; }
  v8f vz = {};
#pragma unroll
  for (int i = 0; i < 4; ++i) accO[i] = vz;

  int ntiles = qt + 1;   // causal: key tiles s0 <= query block
  for (int it = 0; it < ntiles; ++it) {
    int s0 = it * 64;
    __syncthreads();
    {  // load V tile transposed into LDS
      int r = tid >> 1, half = tid & 1;
      const bf16_t* vrow = Vb + (baseBT + s0 + r) * DD + hoff + half * 32;
#pragma unroll
      for (int c = 0; c < 32; ++c) VT[half * 32 + c][r] = vrow[c];
    }
    __syncthreads();

    // Scores: 16 rows x 64 keys = 4 WMMA col tiles, K fragments from global
    v8f S[4];
#pragma unroll
    for (int st = 0; st < 4; ++st) {
      v8f acc = vz;
      const bf16_t* krow = Kb + (baseBT + s0 + st * 16 + l16) * DD + hoff;
#pragma unroll
      for (int kt = 0; kt < 2; ++kt) {
        v16bf bK = pack16(krow + kt * 32 + grp * 8, krow + kt * 32 + 16 + grp * 8);
        acc = wmma_bf16(aQ[kt], bK, acc);
      }
      S[st] = acc;
    }

    // scale + causal mask + online softmax
    float rm[8];
#pragma unroll
    for (int r = 0; r < 8; ++r) rm[r] = -3.0e38f;
#pragma unroll
    for (int st = 0; st < 4; ++st) {
      int key = s0 + st * 16 + l16;
#pragma unroll
      for (int r = 0; r < 8; ++r) {
        int qrow = qbase + grp * 8 + r;
        float vsc = S[st][r] * 0.125f;
        if (key > qrow) vsc = -1.0e9f;
        S[st][r] = vsc;
        rm[r] = fmaxf(rm[r], vsc);
      }
    }
#pragma unroll
    for (int off = 1; off < 16; off <<= 1)
#pragma unroll
      for (int r = 0; r < 8; ++r) rm[r] = fmaxf(rm[r], __shfl_xor(rm[r], off, 32));

    float alpha[8], rs[8];
#pragma unroll
    for (int r = 0; r < 8; ++r) {
      float mn = fmaxf(mrow[r], rm[r]);
      alpha[r] = expf(mrow[r] - mn);
      mrow[r]  = mn;
      rs[r]    = 0.f;
    }
#pragma unroll
    for (int st = 0; st < 4; ++st)
#pragma unroll
      for (int r = 0; r < 8; ++r) {
        float p = expf(S[st][r] - mrow[r]);
        rs[r] += p;
        Pm[w][grp * 8 + r][st * 16 + l16] = (bf16_t)p;
      }
#pragma unroll
    for (int off = 1; off < 16; off <<= 1)
#pragma unroll
      for (int r = 0; r < 8; ++r) rs[r] += __shfl_xor(rs[r], off, 32);
#pragma unroll
    for (int r = 0; r < 8; ++r) lrow[r] = lrow[r] * alpha[r] + rs[r];
#pragma unroll
    for (int nt = 0; nt < 4; ++nt)
#pragma unroll
      for (int r = 0; r < 8; ++r) accO[nt][r] = accO[nt][r] * alpha[r];

    // P x V  (wave-private P through LDS for A-layout)
    v16bf aP[2];
    {
      const bf16_t* prow = &Pm[w][l16][0];
#pragma unroll
      for (int kt = 0; kt < 2; ++kt)
        aP[kt] = pack16(prow + kt * 32 + grp * 8, prow + kt * 32 + 16 + grp * 8);
    }
#pragma unroll
    for (int nt = 0; nt < 4; ++nt)
#pragma unroll
      for (int kt = 0; kt < 2; ++kt) {
        const bf16_t* vt = &VT[nt * 16 + l16][kt * 32 + grp * 8];
        v16bf bV = pack16(vt, vt + 16);
        accO[nt] = wmma_bf16(aP[kt], bV, accO[nt]);
      }
  }

  // normalize + zero padded query rows + store
  unsigned char qpad[8];
#pragma unroll
  for (int r = 0; r < 8; ++r) qpad[r] = msk[baseBT + qbase + grp * 8 + r];
#pragma unroll
  for (int nt = 0; nt < 4; ++nt)
#pragma unroll
    for (int r = 0; r < 8; ++r) {
      float denom = lrow[r];
      float o = (denom > 0.f && !qpad[r]) ? accO[nt][r] / denom : 0.f;
      long m = baseBT + qbase + grp * 8 + r;
      Y[m * DD + hoff + nt * 16 + l16] = (bf16_t)o;
    }
}

// ---------------------------------------------------------------------------
// Host launcher
// ---------------------------------------------------------------------------
extern "C" void kernel_launch(void* const* d_in, const int* in_sizes, int n_in,
                              void* d_out, int out_size, void* d_ws, size_t ws_size,
                              hipStream_t stream) {
  const float* x          = (const float*)d_in[0];
  const unsigned char* xm = (const unsigned char*)d_in[1];
  const float* pos        = (const float*)d_in[2];
  const float* w_qkv      = (const float*)d_in[3];
  const float* w_o        = (const float*)d_in[4];
  const float* ln1g       = (const float*)d_in[5];
  const float* ln1b       = (const float*)d_in[6];
  const float* ln2g       = (const float*)d_in[7];
  const float* ln2b       = (const float*)d_in[8];
  const float* w1         = (const float*)d_in[9];
  const float* w2         = (const float*)d_in[10];
  float* outp             = (float*)d_out;

  char* ws = (char*)d_ws;
  size_t off = 0;
  auto alloc = [&](size_t bytes) -> char* {
    char* p = ws + off;
    off += (bytes + 255) & ~(size_t)255;
    return p;
  };
  bf16_t* q_in  = (bf16_t*)alloc(4194304ull * 2);          // BT*D bf16
  bf16_t* wqkvB = (bf16_t*)alloc(3145728ull * 2);          // 3072*1024
  bf16_t* woB   = (bf16_t*)alloc(1048576ull * 2);          // 1024*1024
  bf16_t* w1r   = (bf16_t*)alloc(12582912ull * 2);         // 4096*3072
  bf16_t* w2r   = (bf16_t*)alloc(12582912ull * 2);         // 1024*12288
  bf16_t* qkvB  = (bf16_t*)alloc(3ull * 4194304ull * 2);   // Q,K,V
  bf16_t* yB    = (bf16_t*)alloc(4194304ull * 2);          // attention out
  float*  x2    = (float*)alloc(4194304ull * 4);           // x + attn proj
  bf16_t* hp1   = (bf16_t*)alloc(2ull * PADT * DD * 2);    // padded LN2 out
  bf16_t* hp2   = (bf16_t*)alloc(2ull * PADT * FF * 2);    // padded gelu out
  (void)ws_size; (void)in_sizes; (void)n_in; (void)out_size;

  // 1. weight conversion / repack
  cvt_kernel<<<(3145728 + 255) / 256, 256, 0, stream>>>(w_qkv, wqkvB, 3145728ll);
  cvt_kernel<<<(1048576 + 255) / 256, 256, 0, stream>>>(w_o, woB, 1048576ll);
  repack_kernel<<<(12582912 + 255) / 256, 256, 0, stream>>>(w1, w1r, DD, 12582912ll);
  repack_kernel<<<(12582912 + 255) / 256, 256, 0, stream>>>(w2, w2r, FF, 12582912ll);

  // 2. LN1 + positional embedding -> q_in (bf16)
  ln_kernel<<<BT, 256, 0, stream>>>(x, ln1g, ln1b, pos, q_in, 0);

  // 3. QKV projection: [4096,1024] x [3072,1024]^T
  gemm_bf16_kernel<<<dim3(3072 / 128, BT / 128), 256, 0, stream>>>(
      q_in, DD, wqkvB, DD, 0, nullptr, nullptr, qkvB, 0);

  // 4. flash attention
  attn_kernel<<<dim3(TT / 64, HH, 2), 128, 0, stream>>>(
      qkvB, qkvB + QKV_STRIDE, qkvB + 2 * QKV_STRIDE, xm, yB);

  // 5. output projection + residual: x2 = x + y @ w_o^T
  gemm_bf16_kernel<<<dim3(DD / 128, BT / 128), 256, 0, stream>>>(
      yB, DD, woB, DD, 1, x, x2, nullptr, 0);

  // 6. LN2 into left-padded buffer (+ zero the 2 pad rows per batch)
  ln_kernel<<<BT, 256, 0, stream>>>(x2, ln2g, ln2b, nullptr, hp1, 1);
  for (int b = 0; b < 2; ++b) {
    zero_bf16_kernel<<<(2 * DD + 255) / 256, 256, 0, stream>>>(hp1 + (long)b * PADT * DD, 2 * DD);
    zero_bf16_kernel<<<(2 * FF + 255) / 256, 256, 0, stream>>>(hp2 + (long)b * PADT * FF, 2 * FF);
  }

  // 7. conv1 (causal, K=3) as GEMM with contiguous 3-row windows, + GELU
  for (int b = 0; b < 2; ++b) {
    gemm_bf16_kernel<<<dim3(FF / 128, TT / 128), 256, 0, stream>>>(
        hp1 + (long)b * PADT * DD, DD, w1r, 3 * DD, 2, nullptr, nullptr,
        hp2 + ((long)b * PADT + 2) * FF, FF);
  }

  // 8. conv2 as GEMM + residual -> final output
  for (int b = 0; b < 2; ++b) {
    gemm_bf16_kernel<<<dim3(DD / 128, TT / 128), 256, 0, stream>>>(
        hp2 + (long)b * PADT * FF, FF, w2r, 3 * FF, 1,
        x2 + (long)b * TT * DD, outp + (long)b * TT * DD, nullptr, 0);
  }
}
